// MoETaskAttention_79989470921147
// MI455X (gfx1250) — compile-verified, matrix operand order checked by reading, not run on hardware
//
#include <hip/hip_runtime.h>
#include <hip/hip_bf16.h>

typedef __attribute__((ext_vector_type(16))) _Float16 v16h;
typedef __attribute__((ext_vector_type(8)))  _Float16 v8h;
typedef __attribute__((ext_vector_type(8)))  float    v8f;

#define BATCH   8
#define NSEQ    1024
#define BN      8192        // BATCH*NSEQ tokens
#define DIMSZ   768
#define NH      8
#define HD      96
#define NE      24
#define QSCALE  0.10206207261596575f   // 96^-0.5

// ---------------------------------------------------------------------------
// WMMA fragment helpers (CDNA5 wave32, 16x16x32 f16 -> f32).
// Per ISA 7.12.2 the per-lane fragment is two contiguous 8-half (128-bit)
// runs: k in [k0+8*half, +8) and [k0+16+8*half, +8).  With operands stored
// f16 and fragment-major contiguous, a fragment is exactly two b128 loads.
// ---------------------------------------------------------------------------
__device__ __forceinline__ v16h load_frag(const _Float16* p) {
    v8h lo = *(const v8h*)p;
    v8h hi = *(const v8h*)(p + 16);
    v16h r;
#pragma unroll
    for (int i = 0; i < 8; ++i) { r[i] = lo[i]; r[i + 8] = hi[i]; }
    return r;
}

// A fragment (row-major M x K, row = lane&15) or B fragment from transposed
// N-major storage Bt[col][k] (col = lane&15): identical address math.
__device__ __forceinline__ v16h frag(const _Float16* M, int ld, int k0, int lane) {
    return load_frag(M + (size_t)(lane & 15) * ld + k0 + ((lane >> 4) << 3));
}

__device__ __forceinline__ v8f wmma16(v16h a, v16h b, v8f c) {
    return __builtin_amdgcn_wmma_f32_16x16x32_f16(false, a, false, b, (short)0, c,
                                                  false, false);
}

// ---------------------------------------------------------------------------
// K0: zero aux-loss stats
// ---------------------------------------------------------------------------
__global__ void zero_stats_kernel(float* stats) {
    if (threadIdx.x < 64) stats[threadIdx.x] = 0.0f;
}

// ---------------------------------------------------------------------------
// Precision/layout conversion kernels (run once per call; ~10M elements total)
// ---------------------------------------------------------------------------
__global__ void cvt_x_kernel(const float* __restrict__ s, _Float16* __restrict__ d, int n) {
    int i = blockIdx.x * blockDim.x + threadIdx.x;
    if (i < n) d[i] = (_Float16)s[i];
}
// W1 [NE][DIMSZ][HD] -> W1t [NE][HD][DIMSZ]
__global__ void cvt_w1_kernel(const float* __restrict__ W1, _Float16* __restrict__ W1t) {
    int i = blockIdx.x * blockDim.x + threadIdx.x;
    if (i >= NE * DIMSZ * HD) return;
    int e = i / (DIMSZ * HD), r = i % (DIMSZ * HD);
    int dd = r / HD, h = r % HD;
    W1t[((size_t)e * HD + h) * DIMSZ + dd] = (_Float16)W1[i];
}
// W2 [NE][HD][DIMSZ] -> W2t [NE][DIMSZ][HD]
__global__ void cvt_w2_kernel(const float* __restrict__ W2, _Float16* __restrict__ W2t) {
    int i = blockIdx.x * blockDim.x + threadIdx.x;
    if (i >= NE * HD * DIMSZ) return;
    int e = i / (HD * DIMSZ), r = i % (HD * DIMSZ);
    int k = r / DIMSZ, c = r % DIMSZ;
    W2t[((size_t)e * DIMSZ + c) * HD + k] = (_Float16)W2[i];
}
// Wkv [DIMSZ][192] -> Wkvt [192][DIMSZ]
__global__ void cvt_wkv_kernel(const float* __restrict__ Wkv, _Float16* __restrict__ Wkvt) {
    int i = blockIdx.x * blockDim.x + threadIdx.x;
    if (i >= DIMSZ * 2 * HD) return;
    int dd = i / (2 * HD), j = i % (2 * HD);
    Wkvt[(size_t)j * DIMSZ + dd] = (_Float16)Wkv[i];
}

// ---------------------------------------------------------------------------
// K1: gating — one wave per token.  stats: [0..23]=me, [24..47]=freq, [48]=z
// ---------------------------------------------------------------------------
__global__ void gate_kernel(const float* __restrict__ x,
                            const float* __restrict__ Wg,
                            const int* __restrict__ task_p,
                            float* __restrict__ gates,
                            int* __restrict__ topi,
                            float* __restrict__ stats) {
    int tok  = blockIdx.x * (blockDim.x >> 5) + (threadIdx.x >> 5);
    int lane = threadIdx.x & 31;
    if (tok >= BN) return;

    const float* Wgt = Wg + (size_t)(*task_p) * DIMSZ * NE;
    const float* xr  = x + (size_t)tok * DIMSZ;

    float acc[NE];
#pragma unroll
    for (int e = 0; e < NE; ++e) acc[e] = 0.0f;
    for (int d = lane; d < DIMSZ; d += 32) {
        float xv = xr[d];
        const float* wr = Wgt + (size_t)d * NE;
#pragma unroll
        for (int e = 0; e < NE; ++e) acc[e] += xv * wr[e];
    }
#pragma unroll
    for (int e = 0; e < NE; ++e) {
        float v = acc[e];
        for (int o = 16; o > 0; o >>= 1) v += __shfl_xor(v, o, 32);
        acc[e] = v;
    }

    if (lane == 0) {
        float m = -1e30f;
#pragma unroll
        for (int e = 0; e < NE; ++e) m = fmaxf(m, acc[e]);
        float p[NE], s = 0.0f;
#pragma unroll
        for (int e = 0; e < NE; ++e) { p[e] = __expf(acc[e] - m); s += p[e]; }
        float inv = 1.0f / s;
#pragma unroll
        for (int e = 0; e < NE; ++e) p[e] *= inv;
        float lse = m + __logf(s);

        bool used[NE];
#pragma unroll
        for (int e = 0; e < NE; ++e) used[e] = false;
        int sel[NH]; float gv[NH]; float gs = 0.0f;
        for (int h = 0; h < NH; ++h) {
            int best = 0; float bv = -1e30f;
            for (int e = 0; e < NE; ++e)
                if (!used[e] && p[e] > bv) { bv = p[e]; best = e; }
            used[best] = true; sel[h] = best; gv[h] = bv; gs += bv;
        }
        float dn = 1.0f / (gs + 1e-6f);
        for (int h = 0; h < NH; ++h) {
            gates[(size_t)tok * NH + h] = gv[h] * dn;
            topi [(size_t)tok * NH + h] = sel[h];
            atomicAdd(&stats[NE + sel[h]], 1.0f);
        }
        for (int e = 0; e < NE; ++e) atomicAdd(&stats[e], p[e]);
        atomicAdd(&stats[2 * NE], lse * lse);
    }
}

// ---------------------------------------------------------------------------
// K2: kv = x @ Wkv + b_kv.  k row-major f16; v stored TRANSPOSED
//     vT[b][d][key] so the attention P.V B-fragment is contiguous.
// ---------------------------------------------------------------------------
__global__ void kv_kernel(const _Float16* __restrict__ xh,
                          const _Float16* __restrict__ Wkvt,
                          const float* __restrict__ bkv,
                          _Float16* __restrict__ kf,
                          _Float16* __restrict__ vT) {
    int lane = threadIdx.x & 31;
    int row0 = blockIdx.x * 16;
    int col0 = blockIdx.y * 16;

    const _Float16* Abase = xh + (size_t)row0 * DIMSZ;
    const _Float16* Bbase = Wkvt + (size_t)col0 * DIMSZ;
    v8f acc = {};
    for (int kc = 0; kc < DIMSZ / 32; ++kc) {
        v16h a = frag(Abase, DIMSZ, kc * 32, lane);
        v16h b = frag(Bbase, DIMSZ, kc * 32, lane);
        acc = wmma16(a, b, acc);
    }
    int col  = col0 + (lane & 15);
    int half = lane >> 4;
    float bias = bkv[col];
#pragma unroll
    for (int i = 0; i < 8; ++i) {
        int t = row0 + i + 8 * half;
        float v = acc[i] + bias;
        if (col < HD)
            kf[(size_t)t * HD + col] = (_Float16)v;
        else
            vT[(size_t)(t >> 10) * HD * NSEQ + (size_t)(col - HD) * NSEQ + (t & (NSEQ - 1))]
                = (_Float16)v;
    }
}

// ---------------------------------------------------------------------------
// K3: dense per-expert q tiles + gather.  Block = (token16, expert), 6 waves.
// ---------------------------------------------------------------------------
__global__ void qmap_kernel(const _Float16* __restrict__ xh,
                            const _Float16* __restrict__ W1t,
                            const int* __restrict__ topi,
                            _Float16* __restrict__ qf) {
    __shared__ int stile[16 * NH];
    int tid  = threadIdx.x;
    int lane = tid & 31;
    int wave = tid >> 5;           // 0..5
    int row0 = blockIdx.x * 16;
    int e    = blockIdx.y;

    if (tid < 16 * NH) stile[tid] = topi[(size_t)row0 * NH + tid];
    __syncthreads();

    const _Float16* Abase = xh  + (size_t)row0 * DIMSZ;
    const _Float16* Bbase = W1t + ((size_t)e * HD + wave * 16) * DIMSZ;

    v8f acc = {};
    for (int kc = 0; kc < DIMSZ / 32; ++kc) {
        v16h a = frag(Abase, DIMSZ, kc * 32, lane);
        v16h b = frag(Bbase, DIMSZ, kc * 32, lane);
        acc = wmma16(a, b, acc);
    }

    int col  = wave * 16 + (lane & 15);
    int half = lane >> 4;
#pragma unroll
    for (int i = 0; i < 8; ++i) {
        int r = i + 8 * half;
        int t = row0 + r;
        float v = acc[i] * QSCALE;
        for (int h = 0; h < NH; ++h)
            if (stile[r * NH + h] == e)
                qf[((size_t)t * NH + h) * HD + col] = (_Float16)v;
    }
}

// ---------------------------------------------------------------------------
// K4: attention per (qtile16, head, batch); 256 threads.
//     LDS: S f32 [16x1024] + P f16 [16x1024] + red [16x16]  (~97KB dynamic)
// ---------------------------------------------------------------------------
__global__ void attn_kernel(const _Float16* __restrict__ qf,
                            const _Float16* __restrict__ kf,
                            const _Float16* __restrict__ vT,
                            _Float16* __restrict__ of) {
    extern __shared__ float smem[];
    float*    S   = smem;                          // 16*1024 f32
    _Float16* Ph  = (_Float16*)(smem + 16 * NSEQ); // 16*1024 f16
    float*    red = (float*)(Ph + 16 * NSEQ);      // 16*16  f32

    int tid  = threadIdx.x;
    int lane = tid & 31;
    int wave = tid >> 5;               // 0..7
    int q0   = blockIdx.x * 16;
    int h    = blockIdx.y;
    int b    = blockIdx.z;

    const _Float16* Qb = qf + ((size_t)(b * NSEQ + q0) * NH + h) * HD;  // ld=NH*HD
    const _Float16* Kb = kf + (size_t)b * NSEQ * HD;
    const _Float16* Vt = vT + (size_t)b * HD * NSEQ;

    // ---- phase 1: S = q.K^T (scale folded into q) ----
    v16h aq[3];
#pragma unroll
    for (int kc = 0; kc < 3; ++kc) aq[kc] = frag(Qb, NH * HD, kc * 32, lane);

    for (int j = wave; j < NSEQ / 16; j += 8) {
        v8f acc = {};
#pragma unroll
        for (int kc = 0; kc < 3; ++kc) {
            v16h bk = frag(Kb + (size_t)j * 16 * HD, HD, kc * 32, lane);
            acc = wmma16(aq[kc], bk, acc);
        }
        int col  = (lane & 15);
        int half = lane >> 4;
#pragma unroll
        for (int i = 0; i < 8; ++i)
            S[(size_t)(i + 8 * half) * NSEQ + j * 16 + col] = acc[i];
    }
    __syncthreads();

    // ---- phase 2: row softmax; emit P as f16 into LDS ----
    int row = tid >> 4, seg = tid & 15;
    float m = -1e30f;
    for (int c = seg * 64; c < seg * 64 + 64; ++c)
        m = fmaxf(m, S[(size_t)row * NSEQ + c]);
    red[row * 16 + seg] = m;
    __syncthreads();
    float rm = -1e30f;
#pragma unroll
    for (int i = 0; i < 16; ++i) rm = fmaxf(rm, red[row * 16 + i]);
    float s = 0.0f;
    for (int c = seg * 64; c < seg * 64 + 64; ++c) {
        float ev = __expf(S[(size_t)row * NSEQ + c] - rm);
        S[(size_t)row * NSEQ + c] = ev;
        s += ev;
    }
    __syncthreads();
    red[row * 16 + seg] = s;
    __syncthreads();
    float tot = 0.0f;
#pragma unroll
    for (int i = 0; i < 16; ++i) tot += red[row * 16 + i];
    float inv = 1.0f / tot;
    for (int c = seg * 64; c < seg * 64 + 64; ++c)
        Ph[(size_t)row * NSEQ + c] = (_Float16)(S[(size_t)row * NSEQ + c] * inv);
    __syncthreads();

    // ---- phase 3: O = P.V (waves 0..5, one 16-col tile each) ----
    if (wave < 6) {
        int c0 = wave * 16;
        v8f acc = {};
        for (int kt = 0; kt < NSEQ / 32; ++kt) {
            v16h a = frag(Ph, NSEQ, kt * 32, lane);              // ds_load_b128
            v16h b = frag(Vt + (size_t)c0 * NSEQ, NSEQ, kt * 32, lane);
            acc = wmma16(a, b, acc);
        }
        int col  = c0 + (lane & 15);
        int half = lane >> 4;
#pragma unroll
        for (int i = 0; i < 8; ++i) {
            int t = b * NSEQ + q0 + i + 8 * half;
            of[((size_t)t * NH + h) * HD + col] = (_Float16)acc[i];
        }
    }
}

// ---------------------------------------------------------------------------
// K5: MoE reduce.  Block per token16 tile (8 waves); loop experts, build
//     gate-weighted mixed[16x96] f16 in LDS, WMMA @ W2t[e], accumulate the
//     full 16x768 out tile in VGPRs (6 col-tiles per wave), single store.
// ---------------------------------------------------------------------------
__global__ void reduce_kernel(const _Float16* __restrict__ of,
                              const _Float16* __restrict__ W2t,
                              const float* __restrict__ gates,
                              const int* __restrict__ topi,
                              float* __restrict__ out) {
    __shared__ __align__(16) _Float16 mixed[16 * HD];
    __shared__ int   stile[16 * NH];
    __shared__ float gtile[16 * NH];

    int tid  = threadIdx.x;
    int lane = tid & 31;
    int wave = tid >> 5;             // col tiles [wave*6, wave*6+6)
    int row0 = blockIdx.x * 16;

    if (tid < 16 * NH) {
        stile[tid] = topi [(size_t)row0 * NH + tid];
        gtile[tid] = gates[(size_t)row0 * NH + tid];
    }

    v8f acc[6];
#pragma unroll
    for (int j = 0; j < 6; ++j) acc[j] = (v8f){};

    for (int e = 0; e < NE; ++e) {
        __syncthreads();             // mixed free to overwrite; stile ready
        if (e + 1 < NE && lane == 0) // warm L2/WGP$ for next expert panel
            __builtin_prefetch(W2t + (size_t)(e + 1) * DIMSZ * HD + wave * 1024, 0, 1);
        for (int idx = tid; idx < 16 * HD; idx += 256) {
            int r = idx / HD, c = idx % HD;
            float sv = 0.0f;
            for (int h = 0; h < NH; ++h)
                if (stile[r * NH + h] == e)
                    sv += gtile[r * NH + h] *
                          (float)of[((size_t)(row0 + r) * NH + h) * HD + c];
            mixed[idx] = (_Float16)sv;
        }
        __syncthreads();

        const _Float16* Bbase = W2t + (size_t)e * DIMSZ * HD;
#pragma unroll
        for (int j = 0; j < 6; ++j) {
            int col0 = (wave * 6 + j) * 16;
            for (int kc = 0; kc < 3; ++kc) {
                v16h a = frag(mixed, HD, kc * 32, lane);
                v16h b = frag(Bbase + (size_t)col0 * HD, HD, kc * 32, lane);
                acc[j] = wmma16(a, b, acc[j]);
            }
        }
    }

    int half = lane >> 4;
#pragma unroll
    for (int j = 0; j < 6; ++j) {
        int col = (wave * 6 + j) * 16 + (lane & 15);
#pragma unroll
        for (int i = 0; i < 8; ++i)
            out[(size_t)(row0 + i + 8 * half) * DIMSZ + col] = acc[j][i];
    }
}

// ---------------------------------------------------------------------------
// K6: finalize aux loss
// ---------------------------------------------------------------------------
__global__ void aux_kernel(const float* __restrict__ stats, float* __restrict__ out) {
    if (threadIdx.x == 0 && blockIdx.x == 0) {
        float me_s = 0.0f, fe_s = 0.0f;
        for (int e = 0; e < NE; ++e) { me_s += stats[e]; fe_s += stats[NE + e]; }
        float sw = 0.0f;
        for (int e = 0; e < NE; ++e)
            sw += (stats[e] / me_s) * (stats[NE + e] / fe_s);
        sw *= (float)NE;
        float z = stats[2 * NE] / (float)BN;
        out[(size_t)BN * DIMSZ] = 0.1f * sw + 0.001f * z;
    }
}

// ---------------------------------------------------------------------------
extern "C" void kernel_launch(void* const* d_in, const int* in_sizes, int n_in,
                              void* d_out, int out_size, void* d_ws, size_t ws_size,
                              hipStream_t stream) {
    const float* x    = (const float*)d_in[0];
    const float* Wg   = (const float*)d_in[1];
    const float* W1   = (const float*)d_in[2];
    const float* W2   = (const float*)d_in[3];
    const float* Wkv  = (const float*)d_in[4];
    const float* bkv  = (const float*)d_in[5];
    const int*   task = (const int*)  d_in[6];
    float* out = (float*)d_out;

    char* ws = (char*)d_ws;
    float*    gates = (float*)    (ws + 0);              // 8192*8 f32
    int*      topi  = (int*)      (ws + 262144);         // 8192*8 i32
    _Float16* xh    = (_Float16*) (ws + 524288);         // 8192*768 f16
    _Float16* qf    = (_Float16*) (ws + 13107200);       // 8192*8*96 f16
    _Float16* kf    = (_Float16*) (ws + 25690112);       // 8192*96 f16
    _Float16* vT    = (_Float16*) (ws + 27262976);       // 8*96*1024 f16
    _Float16* of    = (_Float16*) (ws + 28835840);       // 8192*8*96 f16
    _Float16* W1t   = (_Float16*) (ws + 41418752);       // 24*96*768 f16
    _Float16* W2t   = (_Float16*) (ws + 44957696);       // 24*768*96 f16
    _Float16* Wkvt  = (_Float16*) (ws + 48496640);       // 192*768 f16
    float*    stats = (float*)    (ws + 48791552);       // 64 f32

    zero_stats_kernel<<<1, 64, 0, stream>>>(stats);
    cvt_x_kernel  <<<(BN * DIMSZ + 255) / 256, 256, 0, stream>>>(x, xh, BN * DIMSZ);
    cvt_w1_kernel <<<(NE * DIMSZ * HD + 255) / 256, 256, 0, stream>>>(W1, W1t);
    cvt_w2_kernel <<<(NE * HD * DIMSZ + 255) / 256, 256, 0, stream>>>(W2, W2t);
    cvt_wkv_kernel<<<(DIMSZ * 2 * HD + 255) / 256, 256, 0, stream>>>(Wkv, Wkvt);

    gate_kernel<<<BN / 4, 128, 0, stream>>>(x, Wg, task, gates, topi, stats);
    kv_kernel<<<dim3(BN / 16, (2 * HD) / 16), 32, 0, stream>>>(xh, Wkvt, bkv, kf, vT);
    qmap_kernel<<<dim3(BN / 16, NE), 192, 0, stream>>>(xh, W1t, topi, qf);
    attn_kernel<<<dim3(NSEQ / 16, NH, BATCH), 256,
                  16 * NSEQ * sizeof(float) + 16 * NSEQ * sizeof(_Float16) +
                  256 * sizeof(float), stream>>>(qf, kf, vT, of);
    reduce_kernel<<<BN / 16, 256, 0, stream>>>(of, W2t, gates, topi, out);
    aux_kernel<<<1, 1, 0, stream>>>(stats, out);
}
